// Qwen3Attention_14491219657185
// MI455X (gfx1250) — matrix operand order repaired and checked by reading, not compiled, not run on hardware
//
#include <hip/hip_runtime.h>

// ---------------- problem constants (match reference) ----------------
#define B_    2
#define S_    2048
#define HID_  2048
#define NH_   32
#define NKV_  8
#define HD_   128
#define MROWS (B_*S_)
#define SCALE_ 0.08838834764831845f  // 1/sqrt(128)

typedef __attribute__((ext_vector_type(16))) __bf16 v16bf;
typedef __attribute__((ext_vector_type(8)))  float  v8f;
typedef __attribute__((ext_vector_type(4)))  int    v4i;

union FragBF { v16bf v; unsigned int u[8]; uint4 q[2]; };
union U16x8  { uint4 q;    unsigned short s[8];  };
union U16x16 { uint4 q[2]; unsigned short s[16]; };

// ---------------- optional gfx1250 async global->LDS path ----------------
#if defined(__has_builtin)
#if __has_builtin(__builtin_amdgcn_global_load_async_to_lds_b128) && \
    __has_builtin(__builtin_amdgcn_s_wait_asynccnt)
#define USE_ASYNC_LDS 1
#endif
#endif

#if USE_ASYNC_LDS
typedef __attribute__((address_space(1))) v4i as1_v4i;  // global int4
typedef __attribute__((address_space(3))) v4i as3_v4i;  // LDS int4
__device__ __forceinline__ void async_b128(const void* g, void* l) {
  __builtin_amdgcn_global_load_async_to_lds_b128(
      (as1_v4i*)(unsigned long long)(uintptr_t)g,
      (as3_v4i*)(unsigned int)(uintptr_t)l, 0, 0);
}
__device__ __forceinline__ void async_wait0() { __builtin_amdgcn_s_wait_asynccnt(0); }
#endif

__device__ __forceinline__ v8f zero8() {
  v8f z;
  #pragma unroll
  for (int i = 0; i < 8; i++) z[i] = 0.f;
  return z;
}

__device__ __forceinline__ unsigned short f32_bf16(float f) {
  unsigned int u = __builtin_bit_cast(unsigned int, f);
  u += 0x7FFFu + ((u >> 16) & 1u);            // round-to-nearest-even
  return (unsigned short)(u >> 16);
}
__device__ __forceinline__ float bf16_f32(unsigned short h) {
  unsigned int u = ((unsigned int)h) << 16;
  return __builtin_bit_cast(float, u);
}

// ---------------- fp32 -> bf16 conversion ----------------
__global__ __launch_bounds__(256) void convert_f32_bf16(
    const float* __restrict__ in, unsigned short* __restrict__ out, size_t n) {
  size_t i = ((size_t)blockIdx.x * 256 + threadIdx.x) * 4;
  #pragma unroll
  for (int j = 0; j < 4; j++)
    if (i + j < n) out[i + j] = f32_bf16(in[i + j]);
}

// ---------------- tiled bf16 WMMA GEMM ----------------
// C[M,N] = A[M,K] * Bw[K,N] + bias
// mode 0: fp32 row-major out.  mode 1: bf16 out scattered to [b, head, s, d].
#define GT_M 64
#define GT_N 128
#define GT_K 32

__global__ __launch_bounds__(256) void gemm_bf16_wmma(
    const unsigned short* __restrict__ A, const unsigned short* __restrict__ Bw,
    const float* __restrict__ bias, void* __restrict__ Cout,
    int M, int N, int K, int mode) {
  __shared__ __align__(16) unsigned short As[GT_M * GT_K];      // [m][k] row-major
  __shared__ __align__(16) unsigned int   Bpt[GT_N * (GT_K/2)]; // [n][k/2] pair-packed

  const int tid  = threadIdx.x;
  const int wave = tid >> 5, lane = tid & 31;
  const int nl = lane & 15, hl = lane >> 4;
  const int wm = wave >> 2, wn = wave & 3;   // 2 x 4 wave grid
  const int rowBase = blockIdx.y * GT_M;
  const int colBase = blockIdx.x * GT_N;

  v8f acc[2][2];
  for (int i = 0; i < 2; i++)
    for (int j = 0; j < 2; j++) acc[i][j] = zero8();

  // staging thread roles
  const int ar  = tid >> 2, ac4 = tid & 3;        // A: one uint4 per thread
  const int bp  = tid >> 4, bn0 = (tid & 15) * 8; // B: k-pair bp, 8 columns

  for (int k0 = 0; k0 < K; k0 += GT_K) {
    __syncthreads();
    // ---- stage A tile 64x32 (contiguous b128 per thread) ----
#if USE_ASYNC_LDS
    async_b128(&A[(size_t)(rowBase + ar) * K + k0 + ac4 * 8], &As[ar * GT_K + ac4 * 8]);
#else
    *(uint4*)&As[ar * GT_K + ac4 * 8] =
        *(const uint4*)&A[(size_t)(rowBase + ar) * K + k0 + ac4 * 8];
#endif
    // ---- stage B tile pair-packed transposed: Bpt[n][p] = Bw[k0+2p][n] | Bw[k0+2p+1][n]<<16
    {
      U16x8 r0, r1;
      r0.q = *(const uint4*)&Bw[(size_t)(k0 + 2 * bp)     * N + colBase + bn0];
      r1.q = *(const uint4*)&Bw[(size_t)(k0 + 2 * bp + 1) * N + colBase + bn0];
      #pragma unroll
      for (int i = 0; i < 8; i++)
        Bpt[(bn0 + i) * (GT_K/2) + bp] =
            (unsigned int)r0.s[i] | ((unsigned int)r1.s[i] << 16);
    }
#if USE_ASYNC_LDS
    async_wait0();
#endif
    __syncthreads();

    // ---- fragments ----
    FragBF af[2], bfr[2];
    #pragma unroll
    for (int i = 0; i < 2; i++) {
      const unsigned short* arow = &As[(wm * 32 + i * 16 + nl) * GT_K + hl * 8];
      af[i].q[0] = *(const uint4*)arow;          // K pairs 0..3 of this half
      af[i].q[1] = *(const uint4*)(arow + 16);   // K pairs 4..7 (+16 bf16)
    }
    #pragma unroll
    for (int j = 0; j < 2; j++) {
      const unsigned int* brow = &Bpt[(wn * 32 + j * 16 + nl) * (GT_K/2) + hl * 4];
      bfr[j].q[0] = *(const uint4*)brow;
      bfr[j].q[1] = *(const uint4*)(brow + 8);
    }
    #pragma unroll
    for (int i = 0; i < 2; i++)
      #pragma unroll
      for (int j = 0; j < 2; j++)
        acc[i][j] = __builtin_amdgcn_wmma_f32_16x16x32_bf16(
            false, af[i].v, false, bfr[j].v, (short)0, acc[i][j], false, false);
  }

  // epilogue (C-layout: row = r + 8*half, col = lane%16)
  for (int i = 0; i < 2; i++)
    for (int j = 0; j < 2; j++) {
      int col = colBase + wn * 32 + j * 16 + nl;
      float bv = bias ? bias[col] : 0.f;
      #pragma unroll
      for (int r = 0; r < 8; r++) {
        int row = rowBase + wm * 32 + i * 16 + hl * 8 + r;
        float val = acc[i][j][r] + bv;
        if (mode == 0) {
          ((float*)Cout)[(size_t)row * N + col] = val;
        } else {
          int head = col >> 7, d = col & (HD_ - 1);
          int b = row / S_, s = row % S_;
          int nh = N >> 7;
          ((unsigned short*)Cout)[(((size_t)b * nh + head) * S_ + s) * HD_ + d] =
              f32_bf16(val);
        }
      }
    }
}

// ---------------- per-head RMSNorm (rows of 128, in-place bf16) ----------------
__global__ __launch_bounds__(256) void rmsnorm128_bf16(
    unsigned short* __restrict__ x, const float* __restrict__ w, int nrows) {
  int wave = threadIdx.x >> 5, lane = threadIdx.x & 31;
  int row = blockIdx.x * 8 + wave;
  if (row >= nrows) return;
  unsigned short* p = x + (size_t)row * HD_;
  float v[4], ss = 0.f;
  #pragma unroll
  for (int i = 0; i < 4; i++) {
    v[i] = bf16_f32(p[lane + i * 32]);
    ss += v[i] * v[i];
  }
  #pragma unroll
  for (int m = 16; m >= 1; m >>= 1) ss += __shfl_xor(ss, m, 32);
  float inv = rsqrtf(ss * (1.f / HD_) + 1e-6f);
  #pragma unroll
  for (int i = 0; i < 4; i++)
    p[lane + i * 32] = f32_bf16(v[i] * inv * w[lane + i * 32]);
}

// ---------------- causal GQA flash attention, bf16 WMMA ----------------
// Block = 4 waves, each wave owns 16 queries (64 consecutive queries per block).
// K chunk row-major in LDS (B-frags are contiguous b128 reads).
// V chunk stored pair-packed transposed: Vpt[d][p] = V[2p][d] | V[2p+1][d]<<16.
__global__ __launch_bounds__(128) void attn_flash_wmma(
    const unsigned short* __restrict__ qbf, const unsigned short* __restrict__ kbf,
    const unsigned short* __restrict__ vbf, unsigned short* __restrict__ obf) {
  __shared__ __align__(16) unsigned short Ks[32 * HD_];
  __shared__ __align__(16) unsigned int   Vpt[HD_ * 16];
  __shared__ __align__(16) unsigned short Pw[4][16 * 32];

  const int tid  = threadIdx.x;
  const int wave = tid >> 5, lane = tid & 31;
  const int nl = lane & 15, hl = lane >> 4;

  const int qb_per_bh = S_ / 64;
  const int bh   = blockIdx.x / qb_per_bh;
  const int qblk = blockIdx.x % qb_per_bh;
  const int b = bh / NH_, h = bh % NH_;
  const int kvh = h / (NH_ / NKV_);
  const int q0 = qblk * 64 + wave * 16;

  // Q fragments (A-layout), 4 K-chunks of 32 covering HD=128
  const unsigned short* qrow = qbf + (((size_t)b * NH_ + h) * S_ + q0) * HD_;
  FragBF qa[4];
  #pragma unroll
  for (int c = 0; c < 4; c++) {
    const unsigned short* qp = &qrow[(size_t)nl * HD_ + c * 32 + hl * 8];
    qa[c].q[0] = *(const uint4*)qp;
    qa[c].q[1] = *(const uint4*)(qp + 16);
  }

  v8f oacc[8];
  #pragma unroll
  for (int t = 0; t < 8; t++) oacc[t] = zero8();
  float rowmax[8], rowsum[8];
  #pragma unroll
  for (int r = 0; r < 8; r++) { rowmax[r] = -3.0e38f; rowsum[r] = 0.f; }

  const unsigned short* kbase = kbf + ((size_t)b * NKV_ + kvh) * S_ * HD_;
  const unsigned short* vbase = vbf + ((size_t)b * NKV_ + kvh) * S_ * HD_;

  // staging roles
  const int vp  = tid >> 3;            // 0..15 : key pair
  const int vd0 = (tid & 7) * 16;      // 16 d-columns per thread

  const int nchunks = qblk * 2 + 2;  // keys 0 .. q0_block+63
  for (int c = 0; c < nchunks; c++) {
    const int k0 = c * 32;
    __syncthreads();
    {
      // K chunk: contiguous 8KB copy into row-major Ks
#if USE_ASYNC_LDS
      #pragma unroll
      for (int i = 0; i < 4; i++)
        async_b128(kbase + (size_t)k0 * HD_ + (tid + i * 128) * 8,
                   (unsigned short*)Ks + (tid + i * 128) * 8);
#else
      const uint4* ksrc = (const uint4*)(kbase + (size_t)k0 * HD_);
      uint4* kd = (uint4*)Ks;
      #pragma unroll
      for (int i = 0; i < 4; i++) kd[tid + i * 128] = ksrc[tid + i * 128];
#endif
      // V chunk: pair-pack + transpose into Vpt
      const unsigned short* r0 = vbase + (size_t)(k0 + 2 * vp)     * HD_ + vd0;
      const unsigned short* r1 = vbase + (size_t)(k0 + 2 * vp + 1) * HD_ + vd0;
      U16x16 ra, rb;
      ra.q[0] = *(const uint4*)r0;  ra.q[1] = *(const uint4*)(r0 + 8);
      rb.q[0] = *(const uint4*)r1;  rb.q[1] = *(const uint4*)(r1 + 8);
      #pragma unroll
      for (int i = 0; i < 16; i++)
        Vpt[(vd0 + i) * 16 + vp] =
            (unsigned int)ra.s[i] | ((unsigned int)rb.s[i] << 16);

      if (c + 1 < nchunks) {  // gfx1250 global_prefetch_b8 for next chunk
        __builtin_prefetch((const char*)(kbase + (size_t)(k0 + 32) * HD_) + tid * 16, 0, 0);
        __builtin_prefetch((const char*)(vbase + (size_t)(k0 + 32) * HD_) + tid * 16, 0, 0);
      }
    }
#if USE_ASYNC_LDS
    async_wait0();
#endif
    __syncthreads();
    if (k0 > q0 + 15) continue;  // fully beyond this wave's causal horizon

    // S = Q K^T : two 16x16 score tiles, K-dim 128 = 4 chained WMMAs each
    float sc[2][8];
    #pragma unroll
    for (int t = 0; t < 2; t++) {
      v8f s = zero8();
      #pragma unroll
      for (int c4 = 0; c4 < 4; c4++) {
        FragBF kb;
        const unsigned short* krow = &Ks[(t * 16 + nl) * HD_ + c4 * 32 + hl * 8];
        kb.q[0] = *(const uint4*)krow;
        kb.q[1] = *(const uint4*)(krow + 16);
        s = __builtin_amdgcn_wmma_f32_16x16x32_bf16(
            false, qa[c4].v, false, kb.v, (short)0, s, false, false);
      }
      int key = k0 + t * 16 + nl;
      #pragma unroll
      for (int r = 0; r < 8; r++) {
        int q = q0 + hl * 8 + r;
        sc[t][r] = s[r] * SCALE_ + (key <= q ? 0.f : -1.0e9f);
      }
    }

    // online softmax (row reductions across the 16 lanes of each half)
    #pragma unroll
    for (int r = 0; r < 8; r++) {
      float cm = fmaxf(sc[0][r], sc[1][r]);
      for (int m = 8; m >= 1; m >>= 1) cm = fmaxf(cm, __shfl_xor(cm, m, 32));
      float nm  = fmaxf(rowmax[r], cm);
      float fac = __expf(rowmax[r] - nm);
      rowmax[r] = nm;
      float p0 = __expf(sc[0][r] - nm);
      float p1 = __expf(sc[1][r] - nm);
      float ps = p0 + p1;
      for (int m = 8; m >= 1; m >>= 1) ps += __shfl_xor(ps, m, 32);
      rowsum[r] = rowsum[r] * fac + ps;
      #pragma unroll
      for (int t2 = 0; t2 < 8; t2++) oacc[t2][r] *= fac;
      Pw[wave][(hl * 8 + r) * 32 + nl]      = f32_bf16(p0);
      Pw[wave][(hl * 8 + r) * 32 + 16 + nl] = f32_bf16(p1);
    }

    // P (C-layout -> A-layout via per-wave LDS; DS ops are in-order per wave)
    FragBF pa;
    {
      const unsigned short* pp = &Pw[wave][nl * 32 + hl * 8];
      pa.q[0] = *(const uint4*)pp;
      pa.q[1] = *(const uint4*)(pp + 16);
    }
    // O += P V : 8 column tiles of 16 covering HD=128, K-dim = 32 keys
    #pragma unroll
    for (int t2 = 0; t2 < 8; t2++) {
      FragBF vbr;
      const unsigned int* vrow = &Vpt[(t2 * 16 + nl) * 16 + hl * 4];
      vbr.q[0] = *(const uint4*)vrow;
      vbr.q[1] = *(const uint4*)(vrow + 8);
      oacc[t2] = __builtin_amdgcn_wmma_f32_16x16x32_bf16(
          false, pa.v, false, vbr.v, (short)0, oacc[t2], false, false);
    }
  }

  // write O / l  as bf16 into [b, s, h*d] for the output GEMM
  #pragma unroll
  for (int t2 = 0; t2 < 8; t2++)
    #pragma unroll
    for (int r = 0; r < 8; r++) {
      int row = b * S_ + q0 + hl * 8 + r;
      int col = h * HD_ + t2 * 16 + nl;
      obf[(size_t)row * (NH_ * HD_) + col] = f32_bf16(oacc[t2][r] / rowsum[r]);
    }
}

// ---------------- host orchestration ----------------
extern "C" void kernel_launch(void* const* d_in, const int* in_sizes, int n_in,
                              void* d_out, int out_size, void* d_ws, size_t ws_size,
                              hipStream_t stream) {
  (void)in_sizes; (void)n_in; (void)out_size; (void)ws_size;
  const float* hidden = (const float*)d_in[0];
  // d_in[1] = attention_mask (pure causal; applied analytically)
  const float* wq = (const float*)d_in[2];
  const float* bq = (const float*)d_in[3];
  const float* wk = (const float*)d_in[4];
  const float* bk = (const float*)d_in[5];
  const float* wv = (const float*)d_in[6];
  const float* bv = (const float*)d_in[7];
  const float* wo = (const float*)d_in[8];
  const float* qn = (const float*)d_in[9];
  const float* kn = (const float*)d_in[10];
  float* out = (float*)d_out;

  char* ws = (char*)d_ws;
  size_t off = 0;
  auto alloc = [&](size_t elems) -> unsigned short* {
    unsigned short* p = (unsigned short*)(ws + off);
    off += ((elems * 2 + 255) & ~(size_t)255);
    return p;
  };
  const int M = MROWS;                 // 4096
  const int NQ = NH_ * HD_;            // 4096
  const int NK = NKV_ * HD_;           // 1024
  unsigned short* hbf = alloc((size_t)M * HID_);
  unsigned short* wqb = alloc((size_t)HID_ * NQ);
  unsigned short* wkb = alloc((size_t)HID_ * NK);
  unsigned short* wvb = alloc((size_t)HID_ * NK);
  unsigned short* wob = alloc((size_t)NQ * HID_);
  unsigned short* qb  = alloc((size_t)B_ * NH_ * S_ * HD_);
  unsigned short* kb  = alloc((size_t)B_ * NKV_ * S_ * HD_);
  unsigned short* vb  = alloc((size_t)B_ * NKV_ * S_ * HD_);
  unsigned short* ab  = alloc((size_t)M * NQ);

  // 1) fp32 -> bf16 conversions
  size_t nh_ = (size_t)M * HID_;
  convert_f32_bf16<<<dim3((unsigned)((nh_ + 1023) / 1024)), 256, 0, stream>>>(hidden, hbf, nh_);
  size_t nwq = (size_t)HID_ * NQ;
  convert_f32_bf16<<<dim3((unsigned)((nwq + 1023) / 1024)), 256, 0, stream>>>(wq, wqb, nwq);
  size_t nwk = (size_t)HID_ * NK;
  convert_f32_bf16<<<dim3((unsigned)((nwk + 1023) / 1024)), 256, 0, stream>>>(wk, wkb, nwk);
  convert_f32_bf16<<<dim3((unsigned)((nwk + 1023) / 1024)), 256, 0, stream>>>(wv, wvb, nwk);
  size_t nwo = (size_t)NQ * HID_;
  convert_f32_bf16<<<dim3((unsigned)((nwo + 1023) / 1024)), 256, 0, stream>>>(wo, wob, nwo);

  // 2) QKV projections (bias fused, scattered to [b, head, s, d] bf16)
  gemm_bf16_wmma<<<dim3(NQ / GT_N, M / GT_M), 256, 0, stream>>>(hbf, wqb, bq, qb, M, NQ, HID_, 1);
  gemm_bf16_wmma<<<dim3(NK / GT_N, M / GT_M), 256, 0, stream>>>(hbf, wkb, bk, kb, M, NK, HID_, 1);
  gemm_bf16_wmma<<<dim3(NK / GT_N, M / GT_M), 256, 0, stream>>>(hbf, wvb, bv, vb, M, NK, HID_, 1);

  // 3) per-head RMSNorm on Q and K (in place)
  int rows_q = B_ * NH_ * S_;
  int rows_k = B_ * NKV_ * S_;
  rmsnorm128_bf16<<<dim3(rows_q / 8), 256, 0, stream>>>(qb, qn, rows_q);
  rmsnorm128_bf16<<<dim3(rows_k / 8), 256, 0, stream>>>(kb, kn, rows_k);

  // 4) causal GQA flash attention
  attn_flash_wmma<<<dim3(B_ * NH_ * (S_ / 64)), 128, 0, stream>>>(qb, kb, vb, ab);

  // 5) output projection -> fp32
  gemm_bf16_wmma<<<dim3(HID_ / GT_N, M / GT_M), 256, 0, stream>>>(ab, wob, nullptr, out, M, HID_, NQ, 0);
}
